// GGNN_27238682592073
// MI455X (gfx1250) — matrix operand order repaired and checked by reading, not compiled, not run on hardware
//
#include <hip/hip_runtime.h>
#include <hip/hip_bf16.h>
#include <math.h>

// Problem constants (match reference)
#define B_ 4
#define N_ 2048
#define E_ 2
#define D_ 32
#define H_ 64
#define C_ 10
#define STEPS_ 5
#define K_ (E_ * N_)   /* 4096 = inner dim of A @ ins */
#define ATOT (2 * K_)  /* 8192 = A last-dim (A_in | A_out) */

typedef __attribute__((ext_vector_type(16))) __bf16 v16bf;
typedef __attribute__((ext_vector_type(8)))  float  v8f;

// round-to-nearest-even f32 -> bf16 (A is exactly {0,1} so lossless for A)
static __device__ __forceinline__ unsigned short f2bf(float f) {
  unsigned u = __float_as_uint(f);
  u += 0x7FFFu + ((u >> 16) & 1u);
  return (unsigned short)(u >> 16);
}

// ---------------------------------------------------------------------------
// 1) One-time: convert A (B,N,8192) f32 -> bf16 (134 MB, fits 192MB L2)
// ---------------------------------------------------------------------------
__global__ void __launch_bounds__(256)
cvtA_kernel(const float* __restrict__ A, unsigned short* __restrict__ Abf, long long n4) {
  long long i = (long long)blockIdx.x * blockDim.x + threadIdx.x;
  long long stride = (long long)gridDim.x * blockDim.x;
  const float4* A4 = (const float4*)A;
  ushort4* O4 = (ushort4*)Abf;
  for (; i < n4; i += stride) {
    float4 v = A4[i];
    ushort4 o;
    o.x = f2bf(v.x); o.y = f2bf(v.y); o.z = f2bf(v.z); o.w = f2bf(v.w);
    O4[i] = o;
  }
}

// ---------------------------------------------------------------------------
// 2) Expert transform: ins/outs[b,e,n,d] = state[b,n,:] @ W[e] + bias[e]
//    Stored TRANSPOSED as (B, D=32, E*N=4096) bf16 so WMMA B-fragments are
//    two contiguous b128 loads per lane. Lane = node-in-group => coalesced
//    stores along m = e*N + n.
// ---------------------------------------------------------------------------
__global__ void __launch_bounds__(128)
expert_kernel(const float* __restrict__ state,
              const float* __restrict__ W_in, const float* __restrict__ b_in,
              const float* __restrict__ W_out, const float* __restrict__ b_out,
              unsigned short* __restrict__ insT, unsigned short* __restrict__ outsT) {
  int wave = threadIdx.x >> 5;
  int lane = threadIdx.x & 31;
  int grp  = blockIdx.x * 4 + wave;   // 0..63 groups of 32 nodes
  int b    = blockIdx.y;
  int e    = blockIdx.z;
  int n    = grp * 32 + lane;

  const float* srow = state + ((long long)b * N_ + n) * D_;
  float s[32];
#pragma unroll
  for (int i = 0; i < 32; ++i) s[i] = srow[i];

  const float* Wi = W_in  + e * 32 * 32;
  const float* Wo = W_out + e * 32 * 32;
  const float* bi = b_in  + e * 32;
  const float* bo = b_out + e * 32;

  for (int d = 0; d < 32; ++d) {
    float ai = bi[d], ao = bo[d];
#pragma unroll
    for (int i = 0; i < 32; ++i) {
      ai = fmaf(s[i], Wi[i * 32 + d], ai);
      ao = fmaf(s[i], Wo[i * 32 + d], ao);
    }
    long long idx = ((long long)b * 32 + d) * K_ + (long long)e * N_ + n;
    insT[idx]  = f2bf(ai);
    outsT[idx] = f2bf(ao);
  }
}

// ---------------------------------------------------------------------------
// 3) Big matmul via WMMA bf16: a_{in|out}[b] = A_{half}[b] @ ins/outs[b]
//    One wave owns TWO 16-row M-tiles across the full N=32 (4 accumulators),
//    so each pair of B fragments feeds 4 WMMAs (load:wmma = 2:1), halving
//    the re-read traffic of the shared ins/outs operand.
//    Block = 128 threads = 4 full waves (EXEC all ones, WMMA requirement).
// ---------------------------------------------------------------------------
__global__ void __launch_bounds__(128)
aggregate_wmma(const unsigned short* __restrict__ Abf,   // (B, N, 8192) bf16
               const unsigned short* __restrict__ insT,  // (B, 32, 4096) bf16
               const unsigned short* __restrict__ outsT,
               float* __restrict__ a_in, float* __restrict__ a_out) {
  int wave  = threadIdx.x >> 5;
  int lane  = threadIdx.x & 31;
  int mpair = blockIdx.x * 4 + wave;   // 0..63 ; each covers rows [mpair*32, +32)
  int b     = blockIdx.y;
  int half  = blockIdx.z;              // 0: A_in@ins, 1: A_out@outs

  const unsigned short* Bsrc = half ? outsT : insT;
  float* Odst = half ? a_out : a_in;

  int m0  = mpair * 32;
  int lhi = lane >> 4;    // 0 or 1
  int llo = lane & 15;

  // A fragments: lane<16 holds K {0..7,16..23}, lane>=16 holds K {8..15,24..31}
  const unsigned short* Arow0 =
      Abf + ((long long)b * N_ + (m0 + llo)) * (long long)ATOT + (long long)half * K_ + lhi * 8;
  const unsigned short* Arow1 = Arow0 + 16LL * ATOT;     // second M-tile, rows +16
  // B fragments: lane<16 holds K {0..7,16..23} at column llo,
  // lane>=16 holds K {8..15,24..31} of the same column.
  const unsigned short* Brow0 = Bsrc + ((long long)b * 32 + llo) * (long long)K_ + lhi * 8;
  const unsigned short* Brow1 = Brow0 + 16LL * K_;       // output columns 16..31

  v8f c00 = {};   // (tile0, n 0..15)
  v8f c01 = {};   // (tile0, n 16..31)
  v8f c10 = {};   // (tile1, n 0..15)
  v8f c11 = {};   // (tile1, n 16..31)

  for (int k0 = 0; k0 < K_; k0 += 32) {
    v16bf a0, a1, bf0, bf1;
    ((uint4*)&a0)[0]  = *(const uint4*)(Arow0 + k0);
    ((uint4*)&a0)[1]  = *(const uint4*)(Arow0 + k0 + 16);
    ((uint4*)&a1)[0]  = *(const uint4*)(Arow1 + k0);
    ((uint4*)&a1)[1]  = *(const uint4*)(Arow1 + k0 + 16);
    ((uint4*)&bf0)[0] = *(const uint4*)(Brow0 + k0);
    ((uint4*)&bf0)[1] = *(const uint4*)(Brow0 + k0 + 16);
    ((uint4*)&bf1)[0] = *(const uint4*)(Brow1 + k0);
    ((uint4*)&bf1)[1] = *(const uint4*)(Brow1 + k0 + 16);

    c00 = __builtin_amdgcn_wmma_f32_16x16x32_bf16(false, a0, false, bf0,
                                                  (short)0, c00, false, false);
    c01 = __builtin_amdgcn_wmma_f32_16x16x32_bf16(false, a0, false, bf1,
                                                  (short)0, c01, false, false);
    c10 = __builtin_amdgcn_wmma_f32_16x16x32_bf16(false, a1, false, bf0,
                                                  (short)0, c10, false, false);
    c11 = __builtin_amdgcn_wmma_f32_16x16x32_bf16(false, a1, false, bf1,
                                                  (short)0, c11, false, false);

    // WGP-scope prefetch of next K-chunk of the (L2-resident) A operand:
    // SCOPE 0 pulls into all cache levels, which is what helps here.
    __builtin_prefetch(Arow0 + k0 + 32, 0, 3);
    __builtin_prefetch(Arow1 + k0 + 32, 0, 3);
  }

  // C layout: VGPR i -> (M = mbase + lhi*8 + i, N = llo) ; second tile N += 16
  float* outB = Odst + (long long)b * N_ * 32;
  int rb0 = m0 + lhi * 8;
  int rb1 = rb0 + 16;
#pragma unroll
  for (int i = 0; i < 8; ++i) {
    outB[(long long)(rb0 + i) * 32 + llo]      = c00[i];
    outB[(long long)(rb0 + i) * 32 + 16 + llo] = c01[i];
    outB[(long long)(rb1 + i) * 32 + llo]      = c10[i];
    outB[(long long)(rb1 + i) * 32 + 16 + llo] = c11[i];
  }
}

// ---------------------------------------------------------------------------
// 4) GRU gates: one wave per node, lane = output dim (D=32 == wave32).
//    a = [a_in, a_out, state] (96) ; r,z = sigmoid(a@W + b);
//    h_hat = tanh([a_in, a_out, r*state] @ Wh + bh); state update.
// ---------------------------------------------------------------------------
__global__ void __launch_bounds__(256)
gru_kernel(const float* __restrict__ a_in, const float* __restrict__ a_out,
           const float* __restrict__ state_in,
           const float* __restrict__ Wr, const float* __restrict__ br,
           const float* __restrict__ Wz, const float* __restrict__ bz,
           const float* __restrict__ Wh, const float* __restrict__ bh,
           float* __restrict__ state_out) {
  int lane = threadIdx.x & 31;
  int wave = threadIdx.x >> 5;
  long long node = (long long)blockIdx.x * 8 + wave;   // 0 .. B*N-1
  long long base = node * 32;

  float ain  = a_in[base + lane];
  float aout = a_out[base + lane];
  float s    = state_in[base + lane];

  float r = br[lane], z = bz[lane];
#pragma unroll 4
  for (int i = 0; i < 32; ++i) {
    float v = __shfl(ain, i, 32);
    r = fmaf(v, Wr[i * 32 + lane], r);
    z = fmaf(v, Wz[i * 32 + lane], z);
  }
#pragma unroll 4
  for (int i = 0; i < 32; ++i) {
    float v = __shfl(aout, i, 32);
    r = fmaf(v, Wr[(32 + i) * 32 + lane], r);
    z = fmaf(v, Wz[(32 + i) * 32 + lane], z);
  }
#pragma unroll 4
  for (int i = 0; i < 32; ++i) {
    float v = __shfl(s, i, 32);
    r = fmaf(v, Wr[(64 + i) * 32 + lane], r);
    z = fmaf(v, Wz[(64 + i) * 32 + lane], z);
  }
  r = 1.f / (1.f + __expf(-r));
  z = 1.f / (1.f + __expf(-z));

  float rs = r * s;
  float h = bh[lane];
#pragma unroll 4
  for (int i = 0; i < 32; ++i) h = fmaf(__shfl(ain,  i, 32), Wh[i * 32 + lane], h);
#pragma unroll 4
  for (int i = 0; i < 32; ++i) h = fmaf(__shfl(aout, i, 32), Wh[(32 + i) * 32 + lane], h);
#pragma unroll 4
  for (int i = 0; i < 32; ++i) h = fmaf(__shfl(rs,   i, 32), Wh[(64 + i) * 32 + lane], h);
  h = tanhf(h);

  state_out[base + lane] = (1.f - z) * s + z * h;
}

// ---------------------------------------------------------------------------
// 5) Head: attention pooling + 2-layer classifier + softmax. One block/batch.
// ---------------------------------------------------------------------------
__global__ void __launch_bounds__(256)
head_kernel(const float* __restrict__ state,
            const float* __restrict__ Wa1, const float* __restrict__ ba1,
            const float* __restrict__ Wa2, const float* __restrict__ ba2,
            const float* __restrict__ Wc1, const float* __restrict__ bc1,
            const float* __restrict__ Wc2, const float* __restrict__ bc2,
            float* __restrict__ out) {
  __shared__ float pool_lds[8][32];
  __shared__ float pooled[32];
  __shared__ float h_lds[64];
  __shared__ float logits[10];

  int b    = blockIdx.x;
  int lane = threadIdx.x & 31;
  int wave = threadIdx.x >> 5;

  float pacc = 0.f;
  for (int n = wave; n < N_; n += 8) {
    const float* srow = state + ((long long)b * N_ + n) * 32;
    float s = srow[lane];
    float t = ba1[lane];
#pragma unroll 4
    for (int i = 0; i < 32; ++i) t = fmaf(__shfl(s, i, 32), Wa1[i * 32 + lane], t);
    t = t > 0.f ? t : 0.01f * t;                 // leaky
    float u = t * Wa2[lane];
    for (int off = 16; off > 0; off >>= 1) u += __shfl_xor(u, off, 32);
    float att = 1.f / (1.f + __expf(-(u + ba2[0])));
    pacc = fmaf(s, att, pacc);
  }
  pool_lds[wave][lane] = pacc;
  __syncthreads();

  if (wave == 0) {
    float p = 0.f;
    for (int w = 0; w < 8; ++w) p += pool_lds[w][lane];
    pooled[lane] = p;
  }
  __syncthreads();

  if (threadIdx.x < 64) {
    int j = threadIdx.x;
    float hs = bc1[j];
    for (int i = 0; i < 32; ++i) hs = fmaf(pooled[i], Wc1[i * 64 + j], hs);
    h_lds[j] = hs > 0.f ? hs : 0.01f * hs;
  }
  __syncthreads();

  if (threadIdx.x < 10) {
    int c = threadIdx.x;
    float l = bc2[c];
    for (int j = 0; j < 64; ++j) l = fmaf(h_lds[j], Wc2[j * 10 + c], l);
    logits[c] = l;
  }
  __syncthreads();

  if (threadIdx.x == 0) {
    float m = logits[0];
    for (int c = 1; c < 10; ++c) m = fmaxf(m, logits[c]);
    float sum = 0.f;
    float ex[10];
    for (int c = 0; c < 10; ++c) { ex[c] = __expf(logits[c] - m); sum += ex[c]; }
    for (int c = 0; c < 10; ++c) out[b * 10 + c] = ex[c] / sum;
  }
}

// ---------------------------------------------------------------------------
extern "C" void kernel_launch(void* const* d_in, const int* in_sizes, int n_in,
                              void* d_out, int out_size, void* d_ws, size_t ws_size,
                              hipStream_t stream) {
  const float* prop  = (const float*)d_in[0];
  const float* A     = (const float*)d_in[1];
  const float* W_in  = (const float*)d_in[2];
  const float* b_in  = (const float*)d_in[3];
  const float* W_out = (const float*)d_in[4];
  const float* b_out = (const float*)d_in[5];
  const float* Wr    = (const float*)d_in[6];
  const float* br_   = (const float*)d_in[7];
  const float* Wz    = (const float*)d_in[8];
  const float* bz_   = (const float*)d_in[9];
  const float* Wh    = (const float*)d_in[10];
  const float* bh_   = (const float*)d_in[11];
  const float* Wa1   = (const float*)d_in[12];
  const float* ba1   = (const float*)d_in[13];
  const float* Wa2   = (const float*)d_in[14];
  const float* ba2   = (const float*)d_in[15];
  const float* Wc1   = (const float*)d_in[16];
  const float* bc1   = (const float*)d_in[17];
  const float* Wc2   = (const float*)d_in[18];
  const float* bc2   = (const float*)d_in[19];
  float* out = (float*)d_out;

  // Workspace carve-up (~141 MB): Abf16 | insT | outsT | a_in | a_out | S0 | S1
  char* ws = (char*)d_ws;
  unsigned short* Abf   = (unsigned short*)ws; ws += (size_t)B_ * N_ * ATOT * 2;
  unsigned short* insT  = (unsigned short*)ws; ws += (size_t)B_ * 32 * K_ * 2;
  unsigned short* outsT = (unsigned short*)ws; ws += (size_t)B_ * 32 * K_ * 2;
  float* a_in  = (float*)ws; ws += (size_t)B_ * N_ * 32 * 4;
  float* a_out = (float*)ws; ws += (size_t)B_ * N_ * 32 * 4;
  float* S0    = (float*)ws; ws += (size_t)B_ * N_ * 32 * 4;
  float* S1    = (float*)ws; ws += (size_t)B_ * N_ * 32 * 4;

  // One-time bf16 conversion of A (binary matrix -> lossless in bf16)
  long long n4 = (long long)B_ * N_ * ATOT / 4;
  cvtA_kernel<<<8192, 256, 0, stream>>>(A, Abf, n4);

  const float* sIn = prop;
  float* sOut = S0;
  for (int step = 0; step < STEPS_; ++step) {
    expert_kernel<<<dim3(16, 4, 2), 128, 0, stream>>>(sIn, W_in, b_in, W_out, b_out,
                                                      insT, outsT);
    aggregate_wmma<<<dim3(16, 4, 2), 128, 0, stream>>>(Abf, insT, outsT, a_in, a_out);
    gru_kernel<<<1024, 256, 0, stream>>>(a_in, a_out, sIn, Wr, br_, Wz, bz_, Wh, bh_,
                                         sOut);
    sIn = sOut;
    sOut = (sOut == S0) ? S1 : S0;
  }

  head_kernel<<<4, 256, 0, stream>>>(sIn, Wa1, ba1, Wa2, ba2, Wc1, bc1, Wc2, bc2, out);
}
